// FranchiseAttention_91164975825293
// MI455X (gfx1250) — matrix operand order repaired and checked by reading, not compiled, not run on hardware
//
#include <hip/hip_runtime.h>
#include <hip/hip_bf16.h>

// ---------------------------------------------------------------------------
// Problem constants (from reference): B=4, S=1024, D=1024, H=16, DK=64, NREL=5
// ---------------------------------------------------------------------------
#define BB   4
#define SS   1024
#define DD   1024
#define HH   16
#define DKK  64
#define MM   (BB * SS)          // 4096 rows for the big GEMMs

typedef __attribute__((ext_vector_type(16))) _Float16 v16h;
typedef __attribute__((ext_vector_type(8)))  _Float16 v8h;
typedef __attribute__((ext_vector_type(8)))  float    v8f;
typedef __attribute__((ext_vector_type(4)))  unsigned tdm_u32x4;
typedef __attribute__((ext_vector_type(8)))  int      tdm_i32x8;
typedef __attribute__((ext_vector_type(4)))  int      tdm_i32x4;

#if defined(__has_builtin)
#if __has_builtin(__builtin_amdgcn_tensor_load_to_lds) && \
    __has_builtin(__builtin_amdgcn_s_wait_tensorcnt)
#define USE_TDM 1
#endif
#endif
#ifndef USE_TDM
#define USE_TDM 0
#endif

__device__ inline v8f vzero8f() {
    v8f z = {0.f, 0.f, 0.f, 0.f, 0.f, 0.f, 0.f, 0.f};
    return z;
}

// WMMA wrapper: D = A(16x32 f16) * B(32x16 f16) + C(16x16 f32)
__device__ inline v8f wmma_f16(v16h a, v16h b, v8f c) {
    return __builtin_amdgcn_wmma_f32_16x16x32_f16(
        /*neg_a=*/false, a, /*neg_b=*/false, b,
        /*c_mod=*/(short)0, c, /*reuse_a=*/false, /*reuse_b=*/false);
}

// A-operand fragment (16x32, row-major source, row stride ldm halves).
// Lane L: row r=L&15, g=L>>4. VGPR0-3: K=8g..8g+7 ; VGPR4-7: K=16+8g..+7
__device__ inline v16h load_frag_a(const _Float16* base, int ldm) {
    int lane = threadIdx.x & 31;
    int r = lane & 15, g = lane >> 4;
    v16h out;
    ((v8h*)&out)[0] = *(const v8h*)(base + (size_t)r * ldm + 8 * g);
    ((v8h*)&out)[1] = *(const v8h*)(base + (size_t)r * ldm + 16 + 8 * g);
    return out;
}

// B-operand fragment (32x16). Source stored transposed: Bt[n][k], stride ldk.
// Lane L: col n=L&15, g=L>>4. VGPR0-3: K=8g..8g+7 ; VGPR4-7: K=16+8g..+7
__device__ inline v16h load_frag_b(const _Float16* base, int ldk) {
    int lane = threadIdx.x & 31;
    int n = lane & 15, g = lane >> 4;
    const _Float16* p = base + (size_t)n * ldk + 8 * g;
    v16h out;
    ((v8h*)&out)[0] = *(const v8h*)(p);
    ((v8h*)&out)[1] = *(const v8h*)(p + 16);
    return out;
}

// Async global->LDS copy of 32 bytes per lane (two B128 ops, ASYNCcnt path).
__device__ inline void async_copy32(const _Float16* gbase, unsigned gOffBytes,
                                    _Float16* ldsDst) {
    unsigned lds = (unsigned)(uintptr_t)ldsDst;         // low 32 bits = LDS addr
    unsigned long long sa = (unsigned long long)(uintptr_t)gbase;
    asm volatile(
        "global_load_async_to_lds_b128 %0, %1, %2 offset:0\n\t"
        "global_load_async_to_lds_b128 %0, %1, %2 offset:16"
        :: "v"(lds), "v"(gOffBytes), "s"(sa)
        : "memory");
}

__device__ inline void wait_async_zero() {
    asm volatile("s_wait_asynccnt 0x0" ::: "memory");
}

#if USE_TDM
// Tensor Data Mover: DMA a 2D tile (tileDim1 rows x tileDim0 elems, 2B elems)
// from a row-major tensor (row stride = stride0 elems) into LDS, packed
// contiguously. D# bit layout per CDNA5 ISA §8.3/8.4.
// This toolchain exposes the 6-arg builtin:
//   (u32x4 g0, i32x8 g1, i32x4 g2, i32x4 g3, i32x8 extra, i32 cpol)
__device__ inline void tdm_load_2d(const _Float16* gtile, void* ldsDst,
                                   unsigned tensorDim0, unsigned tensorDim1,
                                   unsigned tileDim0, unsigned tileDim1,
                                   unsigned stride0) {
    unsigned long long ga = (unsigned long long)(uintptr_t)gtile;
    unsigned lds = (unsigned)(uintptr_t)ldsDst;
    tdm_u32x4 g0;
    g0[0] = 1u;                                            // count=1, user D#
    g0[1] = lds;                                           // lds_addr
    g0[2] = (unsigned)(ga & 0xFFFFFFFFu);                  // global_addr[31:0]
    g0[3] = (unsigned)((ga >> 32) & 0x01FFFFFFu) | (2u << 30); // addr[56:32], type=2
    tdm_i32x8 g1;
    g1[0] = (int)(1u << 16);                               // data_size=1 (2 bytes)
    g1[1] = (int)((tensorDim0 & 0xFFFFu) << 16);           // dim0[15:0] @ bits 63:48
    g1[2] = (int)(((tensorDim0 >> 16) & 0xFFFFu) |
                  ((tensorDim1 & 0xFFFFu) << 16));         // dim0[31:16], dim1[15:0]
    g1[3] = (int)(((tensorDim1 >> 16) & 0xFFFFu) |
                  ((tileDim0 & 0xFFFFu) << 16));           // dim1[31:16], tile_dim0
    g1[4] = (int)(tileDim1 & 0xFFFFu);                     // tile_dim1; tile_dim2=0
    g1[5] = (int)stride0;                                  // dim0_stride[31:0]
    g1[6] = 0;                                             // dim0_stride[47:32]=0
    g1[7] = 0;
    tdm_i32x4 z4 = {0, 0, 0, 0};
    tdm_i32x8 z8 = {0, 0, 0, 0, 0, 0, 0, 0};
    __builtin_amdgcn_tensor_load_to_lds(g0, g1, z4, z4, z8, 0);
}
#endif

// ---------------------------------------------------------------------------
// Elementwise converts / transposes
// ---------------------------------------------------------------------------
__global__ void cvt_f32_to_h(const float* __restrict__ in,
                             _Float16* __restrict__ out, int n) {
    int i = blockIdx.x * blockDim.x + threadIdx.x;
    if (i < n) out[i] = (_Float16)in[i];
}

// Wt[n][k] = (f16) W[k][n] -- LDS-tiled transpose, coalesced both sides.
__global__ __launch_bounds__(256) void transpose_w(const float* __restrict__ W,
                                                   _Float16* __restrict__ Wt) {
    __shared__ _Float16 tile[32][33];
    int n0 = blockIdx.x * 32;
    int k0 = blockIdx.y * 32;
    for (int i = threadIdx.y; i < 32; i += 8)
        tile[i][threadIdx.x] = (_Float16)W[(size_t)(k0 + i) * DD + n0 + threadIdx.x];
    __syncthreads();
    for (int i = threadIdx.y; i < 32; i += 8)
        Wt[(size_t)(n0 + i) * DD + k0 + threadIdx.x] = tile[threadIdx.x][i];
}

// ---------------------------------------------------------------------------
// Tiled WMMA GEMM, double-buffered staging via TDM (or async-to-LDS fallback):
//   C[M,N] = A[M,K] @ Bt[N,K]^T (+ bias) (+ residual)
//   MODE 0: outH (f16) = acc + bias[n]
//   MODE 1: outF (f32) = acc + bias[n] + resid[m,n]
// Block: 256 threads (8 waves, 4x2), tile 128x128, K step 32.
// ---------------------------------------------------------------------------
template <int MODE>
__global__ __launch_bounds__(256) void gemm_wmma(
    const _Float16* __restrict__ A, const _Float16* __restrict__ Bt,
    const float* __restrict__ bias, const float* __restrict__ resid,
    _Float16* __restrict__ outH, float* __restrict__ outF,
    int M, int N, int K)
{
    __shared__ _Float16 As[2][128 * 32];
    __shared__ _Float16 Bs[2][128 * 32];

    int t = threadIdx.x;
    int wave = t >> 5;
    int wm = wave >> 1;          // 0..3
    int wn = wave & 1;           // 0..1
    int m0 = blockIdx.x * 128;
    int n0 = blockIdx.y * 128;

    int lr = t >> 1;             // 0..127
    int lc = (t & 1) * 16;       // 0 or 16
    (void)lr; (void)lc;

    v8f acc[2][4];
#pragma unroll
    for (int mi = 0; mi < 2; ++mi)
#pragma unroll
        for (int ni = 0; ni < 4; ++ni) acc[mi][ni] = vzero8f();

    auto stage = [&](int buf, int k0) {
#if USE_TDM
        if (wave == 0) {
            tdm_load_2d(A + (size_t)m0 * K + k0, &As[buf][0], K, M, 32, 128, K);
            tdm_load_2d(Bt + (size_t)n0 * K + k0, &Bs[buf][0], K, N, 32, 128, K);
        }
#else
        async_copy32(A, (unsigned)(((size_t)(m0 + lr) * K + k0 + lc) * 2),
                     &As[buf][lr * 32 + lc]);
        async_copy32(Bt, (unsigned)(((size_t)(n0 + lr) * K + k0 + lc) * 2),
                     &Bs[buf][lr * 32 + lc]);
#endif
    };
    auto wait_stage = [&]() {
#if USE_TDM
        if (wave == 0) __builtin_amdgcn_s_wait_tensorcnt(0);
#else
        wait_async_zero();
#endif
    };

    stage(0, 0);
    wait_stage();
    __syncthreads();

    for (int k0 = 0; k0 < K; k0 += 32) {
        int buf = (k0 >> 5) & 1;
        if (k0 + 32 < K) stage(buf ^ 1, k0 + 32);   // DMA next tile during compute

        v16h af[2], bf[4];
#pragma unroll
        for (int mi = 0; mi < 2; ++mi)
            af[mi] = load_frag_a(&As[buf][(wm * 32 + mi * 16) * 32], 32);
#pragma unroll
        for (int ni = 0; ni < 4; ++ni)
            bf[ni] = load_frag_b(&Bs[buf][(wn * 64 + ni * 16) * 32], 32);

#pragma unroll
        for (int mi = 0; mi < 2; ++mi)
#pragma unroll
            for (int ni = 0; ni < 4; ++ni)
                acc[mi][ni] = wmma_f16(af[mi], bf[ni], acc[mi][ni]);

        wait_stage();            // next tile landed in the other buffer
        __syncthreads();
    }

    // Epilogue. C layout: lane L -> col n=L&15, rows m = 8*(L>>4)+v
    int lane = t & 31;
    int cn = lane & 15, g = lane >> 4;
#pragma unroll
    for (int mi = 0; mi < 2; ++mi) {
#pragma unroll
        for (int ni = 0; ni < 4; ++ni) {
            int ng = n0 + wn * 64 + ni * 16 + cn;
            int mg = m0 + wm * 32 + mi * 16 + 8 * g;
            float bv = bias[ng];
#pragma unroll
            for (int v = 0; v < 8; ++v) {
                int m = mg + v;
                float val = acc[mi][ni][v] + bv;
                if (MODE == 0) {
                    outH[(size_t)m * N + ng] = (_Float16)val;
                } else {
                    outF[(size_t)m * N + ng] = val + resid[(size_t)m * N + ng];
                }
            }
        }
    }
}

// ---------------------------------------------------------------------------
// Repack Q/K projections into per-head layout [BH, S, 64].
// K is pre-scaled by 1/sqrt(DK) so the attention loop needs no score scaling.
// ---------------------------------------------------------------------------
__global__ void repack_qk(const _Float16* __restrict__ Qf,
                          const _Float16* __restrict__ Kf,
                          _Float16* __restrict__ Qh,
                          _Float16* __restrict__ Kh) {
    int i = blockIdx.x * blockDim.x + threadIdx.x;
    if (i >= BB * SS * DD) return;
    int d = i & (DD - 1);
    int s = (i >> 10) & (SS - 1);
    int b = i >> 20;
    int h = d >> 6, dk = d & 63;
    int bh = b * HH + h;
    size_t dst = ((size_t)bh * SS + s) * DKK + dk;
    Qh[dst] = Qf[i];
    Kh[dst] = (_Float16)((float)Kf[i] * 0.125f);
}

// V -> Vt [BH, 64, S], LDS-tiled transpose. grid (BH, S/32, DK/32), block (32,8)
__global__ __launch_bounds__(256) void transpose_v(const _Float16* __restrict__ Vf,
                                                   _Float16* __restrict__ Vt) {
    __shared__ _Float16 tile[32][33];
    int bh = blockIdx.x;
    int b = bh >> 4, h = bh & 15;
    int s0 = blockIdx.y * 32;
    int dk0 = blockIdx.z * 32;
    for (int i = threadIdx.y; i < 32; i += 8)
        tile[i][threadIdx.x] =
            Vf[((size_t)b * SS + s0 + i) * DD + h * DKK + dk0 + threadIdx.x];
    __syncthreads();
    for (int i = threadIdx.y; i < 32; i += 8)
        Vt[((size_t)bh * DKK + dk0 + i) * SS + s0 + threadIdx.x] = tile[threadIdx.x][i];
}

// ---------------------------------------------------------------------------
// code[b,i,j] = relm[i,j] | (mask[b,i,j] ? 8 : 0)  -- one byte per element
// ---------------------------------------------------------------------------
__global__ void code_kernel(const int* __restrict__ relm,
                            const int* __restrict__ mask,
                            unsigned char* __restrict__ code) {
    int i = blockIdx.x * blockDim.x + threadIdx.x;   // over B*S*S
    if (i >= BB * SS * SS) return;
    int ij = i & (SS * SS - 1);
    unsigned char c = (unsigned char)(relm[ij] & 7);
    if (mask[i] != 0) c |= 8;
    code[i] = c;
}

// ---------------------------------------------------------------------------
// Qrel[bh, s, r] = 0.5 * sum_d Qh[bh,s,d] * rel_table[r,d]  (r < 5, pad to 8)
// ---------------------------------------------------------------------------
__global__ void qrel_kernel(const _Float16* __restrict__ Qh,
                            const float* __restrict__ rel_table,
                            float* __restrict__ Qrel) {
    int idx = blockIdx.x * blockDim.x + threadIdx.x;   // over BH*S
    if (idx >= BB * HH * SS) return;
    const _Float16* q = Qh + (size_t)idx * DKK;
    float acc[5] = {0.f, 0.f, 0.f, 0.f, 0.f};
#pragma unroll 4
    for (int d = 0; d < DKK; ++d) {
        float qv = (float)q[d];
#pragma unroll
        for (int r = 0; r < 5; ++r) acc[r] += qv * rel_table[r * DKK + d];
    }
#pragma unroll
    for (int r = 0; r < 5; ++r) Qrel[(size_t)idx * 8 + r] = 0.5f * acc[r];
    for (int r = 5; r < 8; ++r) Qrel[(size_t)idx * 8 + r] = 0.f;
}

// ---------------------------------------------------------------------------
// Flash attention. One wave (32 thr) per (bh, 16 query rows).
// Per 32-key step: 4 WMMAs for scores, byte-code gather (rel+mask packed) +
// LDS-staged Qrel bias + online softmax, then 4 WMMAs for P@V.
// ---------------------------------------------------------------------------
__global__ __launch_bounds__(32) void attn_kernel(
    const _Float16* __restrict__ Qh,   // [BH,S,64]
    const _Float16* __restrict__ Kh,   // [BH,S,64] (pre-scaled by 1/8)
    const _Float16* __restrict__ Vt,   // [BH,64,S]
    const float* __restrict__ Qrel,    // [BH,S,8] (pre-scaled by 0.5)
    const unsigned char* __restrict__ code, // [B,S,S] rel|mask<<3
    _Float16* __restrict__ ctxOut)     // [B,S,D]
{
    __shared__ _Float16 Pl[16 * 32];
    __shared__ float sQr[16 * 8];

    int bh = blockIdx.y;
    int b = bh >> 4, h = bh & 15;
    int i0 = blockIdx.x * 16;
    int lane = threadIdx.x & 31;
    int n = lane & 15, g = lane >> 4;

    const _Float16* Qp = Qh + ((size_t)bh * SS + i0) * DKK;
    const _Float16* Kp = Kh + (size_t)bh * SS * DKK;
    const _Float16* Vp = Vt + (size_t)bh * DKK * SS;
    const unsigned char* cp = code + (size_t)b * SS * SS;

    // stage this block's Qrel rows (16 x 8 floats) into LDS
    {
        const float4* src4 = (const float4*)(Qrel + ((size_t)bh * SS + i0) * 8);
        ((float4*)sQr)[lane] = src4[lane];
    }
    __syncthreads();

    v16h aq0 = load_frag_a(Qp, DKK);        // dk 0..31
    v16h aq1 = load_frag_a(Qp + 32, DKK);   // dk 32..63

    v8f ctx[4];
#pragma unroll
    for (int tt = 0; tt < 4; ++tt) ctx[tt] = vzero8f();
    float mrow[8], lrow[8];
#pragma unroll
    for (int v = 0; v < 8; ++v) { mrow[v] = -3.0e38f; lrow[v] = 0.f; }

    for (int j = 0; j < SS; j += 32) {
        // prefetch next key/value block while this one computes
        if (j + 32 < SS) {
            __builtin_prefetch(Kp + (size_t)(j + 32 + n) * DKK, 0, 1);
            __builtin_prefetch(Vp + (size_t)n * SS + j + 32, 0, 1);
        }

        // scores for keys [j, j+32)  (K pre-scaled, so this is QK/sqrt(dk))
        v16h bk00 = load_frag_b(Kp + (size_t)j * DKK, DKK);
        v16h bk01 = load_frag_b(Kp + (size_t)j * DKK + 32, DKK);
        v16h bk10 = load_frag_b(Kp + (size_t)(j + 16) * DKK, DKK);
        v16h bk11 = load_frag_b(Kp + (size_t)(j + 16) * DKK + 32, DKK);

        v8f s0 = vzero8f(), s1 = vzero8f();
        s0 = wmma_f16(aq0, bk00, s0);
        s0 = wmma_f16(aq1, bk01, s0);
        s1 = wmma_f16(aq0, bk10, s1);
        s1 = wmma_f16(aq1, bk11, s1);

#pragma unroll
        for (int v = 0; v < 8; ++v) {
            int r = 8 * g + v;                 // row within 16-block
            size_t irow = (size_t)(i0 + r) * SS;
            unsigned c0 = cp[irow + j + n];
            unsigned c1 = cp[irow + j + 16 + n];
            float x0 = s0[v] + sQr[r * 8 + (c0 & 7)];
            float x1 = s1[v] + sQr[r * 8 + (c1 & 7)];
            if (!(c0 & 8)) x0 = -1.0e9f;
            if (!(c1 & 8)) x1 = -1.0e9f;

            float mx = fmaxf(x0, x1);
#pragma unroll
            for (int off = 1; off < 16; off <<= 1)
                mx = fmaxf(mx, __shfl_xor(mx, off, 16));
            float mnew = fmaxf(mrow[v], mx);
            float scale = __expf(mrow[v] - mnew);
            mrow[v] = mnew;
            float p0 = __expf(x0 - mnew);
            float p1 = __expf(x1 - mnew);
            float rs = p0 + p1;
#pragma unroll
            for (int off = 1; off < 16; off <<= 1)
                rs += __shfl_xor(rs, off, 16);
            lrow[v] = lrow[v] * scale + rs;
#pragma unroll
            for (int tt = 0; tt < 4; ++tt) ctx[tt][v] *= scale;
            // stage P (C-layout -> LDS) for A-layout reload
            Pl[r * 32 + n] = (_Float16)p0;
            Pl[r * 32 + 16 + n] = (_Float16)p1;
        }
        __syncthreads();

        v16h pf = load_frag_a(Pl, 32);   // P: 16 rows x 32 keys
#pragma unroll
        for (int tt = 0; tt < 4; ++tt) {
            v16h bv = load_frag_b(Vp + (size_t)tt * 16 * SS + j, SS);
            ctx[tt] = wmma_f16(pf, bv, ctx[tt]);
        }
        __syncthreads();
    }

    // normalize + scatter to [B,S,D] f16
#pragma unroll
    for (int tt = 0; tt < 4; ++tt) {
#pragma unroll
        for (int v = 0; v < 8; ++v) {
            int i = i0 + 8 * g + v;
            int dk = tt * 16 + n;
            float val = ctx[tt][v] / lrow[v];
            ctxOut[((size_t)b * SS + i) * DD + h * DKK + dk] = (_Float16)val;
        }
    }
}

// ---------------------------------------------------------------------------
// LayerNorm over last dim (D=1024), one block per row
// ---------------------------------------------------------------------------
__global__ __launch_bounds__(256) void ln_kernel(
    const float* __restrict__ y, const float* __restrict__ gamma,
    const float* __restrict__ beta, float* __restrict__ out)
{
    __shared__ float red[256];
    int row = blockIdx.x;
    const float* yr = y + (size_t)row * DD;
    int t = threadIdx.x;

    float s = 0.f;
    for (int d = t; d < DD; d += 256) s += yr[d];
    red[t] = s; __syncthreads();
    for (int off = 128; off > 0; off >>= 1) {
        if (t < off) red[t] += red[t + off];
        __syncthreads();
    }
    float mu = red[0] / (float)DD;
    __syncthreads();

    float var = 0.f;
    for (int d = t; d < DD; d += 256) {
        float dv = yr[d] - mu;
        var += dv * dv;
    }
    red[t] = var; __syncthreads();
    for (int off = 128; off > 0; off >>= 1) {
        if (t < off) red[t] += red[t + off];
        __syncthreads();
    }
    float rstd = rsqrtf(red[0] / (float)DD + 1e-5f);

    for (int d = t; d < DD; d += 256)
        out[(size_t)row * DD + d] = (yr[d] - mu) * rstd * gamma[d] + beta[d];
}

// ---------------------------------------------------------------------------
// Host launcher
// ---------------------------------------------------------------------------
extern "C" void kernel_launch(void* const* d_in, const int* in_sizes, int n_in,
                              void* d_out, int out_size, void* d_ws, size_t ws_size,
                              hipStream_t stream) {
    (void)in_sizes; (void)n_in; (void)out_size; (void)ws_size;
    const float* x    = (const float*)d_in[0];
    const int*   relm = (const int*)d_in[1];
    const int*   mask = (const int*)d_in[2];
    const float* Wq = (const float*)d_in[3];  const float* bq = (const float*)d_in[4];
    const float* Wk = (const float*)d_in[5];  const float* bk = (const float*)d_in[6];
    const float* Wv = (const float*)d_in[7];  const float* bv = (const float*)d_in[8];
    const float* Wo = (const float*)d_in[9];  const float* bo = (const float*)d_in[10];
    const float* rel_table = (const float*)d_in[11];
    const float* gamma = (const float*)d_in[12];
    const float* beta  = (const float*)d_in[13];
    float* out = (float*)d_out;

    char* ws = (char*)d_ws;
    const size_t MB = 1024ull * 1024ull;
    // Workspace layout (64 MB total, with safe aliasing by stream order):
    _Float16* xb   = (_Float16*)(ws + 0);        // 8MB; reused as ctx after projections
    _Float16* WqT  = (_Float16*)(ws + 8 * MB);   // 2MB each
    _Float16* WkT  = (_Float16*)(ws + 10 * MB);
    _Float16* WvT  = (_Float16*)(ws + 12 * MB);
    _Float16* WoT  = (_Float16*)(ws + 14 * MB);
    _Float16* Qf   = (_Float16*)(ws + 16 * MB);  // 8MB; [16..32MB) reused as y (f32)
    _Float16* Kf   = (_Float16*)(ws + 24 * MB);
    _Float16* Vf   = (_Float16*)(ws + 32 * MB);  // 8MB; reused as Qrel+code later
    _Float16* Qh   = (_Float16*)(ws + 40 * MB);
    _Float16* Kh   = (_Float16*)(ws + 48 * MB);
    _Float16* Vt   = (_Float16*)(ws + 56 * MB);
    float*    Qrel = (float*)(ws + 32 * MB);         // 2MB (over dead Vf)
    unsigned char* code = (unsigned char*)(ws + 34 * MB); // 4MB (over dead Vf)
    float*    y    = (float*)(ws + 16 * MB);         // 16MB (over dead Qf+Kf)
    _Float16* ctx  = xb;                             // over dead xb

    const int nElem = BB * SS * DD;              // 4M

    // 1) convert x to f16
    cvt_f32_to_h<<<nElem / 256, 256, 0, stream>>>(x, xb, nElem);
    // 2) weights -> transposed f16 (tiled)
    {
        dim3 tgrid(DD / 32, DD / 32), tblk(32, 8);
        transpose_w<<<tgrid, tblk, 0, stream>>>(Wq, WqT);
        transpose_w<<<tgrid, tblk, 0, stream>>>(Wk, WkT);
        transpose_w<<<tgrid, tblk, 0, stream>>>(Wv, WvT);
        transpose_w<<<tgrid, tblk, 0, stream>>>(Wo, WoT);
    }

    // 3) Q/K/V projections (WMMA GEMM, TDM/async double-buffered staging)
    dim3 ggrid(MM / 128, DD / 128);
    gemm_wmma<0><<<ggrid, 256, 0, stream>>>(xb, WqT, bq, nullptr, Qf, nullptr,
                                            MM, DD, DD);
    gemm_wmma<0><<<ggrid, 256, 0, stream>>>(xb, WkT, bk, nullptr, Kf, nullptr,
                                            MM, DD, DD);
    gemm_wmma<0><<<ggrid, 256, 0, stream>>>(xb, WvT, bv, nullptr, Vf, nullptr,
                                            MM, DD, DD);

    // 4) repack into per-head layouts (K pre-scaled; V transposed, tiled)
    repack_qk<<<nElem / 256, 256, 0, stream>>>(Qf, Kf, Qh, Kh);
    {
        dim3 vgrid(BB * HH, SS / 32, DKK / 32), vblk(32, 8);
        transpose_v<<<vgrid, vblk, 0, stream>>>(Vf, Vt);
    }

    // 5) Qrel precompute + packed rel/mask byte codes (after Vf is dead)
    qrel_kernel<<<(BB * HH * SS) / 256, 256, 0, stream>>>(Qh, rel_table, Qrel);
    code_kernel<<<(BB * SS * SS) / 256, 256, 0, stream>>>(relm, mask, code);

    // 6) flash attention
    dim3 agrid(SS / 16, BB * HH);
    attn_kernel<<<agrid, 32, 0, stream>>>(Qh, Kh, Vt, Qrel, code, ctx);

    // 7) output projection + bias + residual (f32)
    gemm_wmma<1><<<ggrid, 256, 0, stream>>>(ctx, WoT, bo, x, nullptr, y,
                                            MM, DD, DD);

    // 8) layernorm -> d_out
    ln_kernel<<<MM, 256, 0, stream>>>(y, gamma, beta, out);
}
